// SimpleMoE_81733227643378
// MI455X (gfx1250) — matrix operand order repaired and checked by reading, not compiled, not run on hardware
//
#include <hip/hip_runtime.h>

// ---------------- problem constants ----------------
#define N_TOK 16384
#define DIM   1024
#define NEXP  8
#define DFF   4096
#define CAP   (N_TOK * 2)   // max (token,slot) entries per expert

// ---------------- CDNA5 WMMA / TDM types ----------------
typedef __attribute__((ext_vector_type(16))) __bf16 bf16x16;
typedef __attribute__((ext_vector_type(8)))  float  f32x8;
typedef unsigned int u32x4 __attribute__((ext_vector_type(4)));
typedef int          i32x4 __attribute__((ext_vector_type(4)));
typedef int          i32x8 __attribute__((ext_vector_type(8)));

union FragAB { bf16x16 v; uint4 q[2]; };

__device__ __forceinline__ unsigned short f2bf(float f) {
  union { float f; unsigned int u; } c; c.f = f;
  unsigned int r = 0x7FFFu + ((c.u >> 16) & 1u);   // round-to-nearest-even
  return (unsigned short)((c.u + r) >> 16);
}

// ---- TDM: one 2D tile (bf16) global->LDS via Tensor Data Mover (D# per ISA 8.3/8.4)
__device__ __forceinline__ void tdm_load_2d(unsigned lds_addr, const void* gptr,
                                            unsigned td0, unsigned td1,
                                            unsigned tile0, unsigned tile1,
                                            unsigned long long stride0,
                                            unsigned padIntervalCode,
                                            unsigned padAmountCode) {
  unsigned long long ga = (unsigned long long)(size_t)gptr;
  u32x4 g0;
  g0[0] = 1u;                                       // count=1 (valid user descriptor)
  g0[1] = lds_addr;                                 // LDS byte address
  g0[2] = (unsigned)(ga & 0xFFFFFFFFull);           // global_addr[31:0]  (bits 95:64)
  g0[3] = (unsigned)((ga >> 32) & 0x01FFFFFFull)    // global_addr[56:32] (bits 120:96)
        | (2u << 30);                               // type=2 ("image")
  i32x8 g1;
  g1[0] = (int)((1u << 16)                          // data_size = 2 bytes
              | (1u << 20)                          // pad_enable
              | (padIntervalCode << 22)
              | (padAmountCode  << 25));
  g1[1] = (int)((td0 & 0xFFFFu) << 16);             // tensor_dim0[15:0] @ 63:48
  g1[2] = (int)((td0 >> 16) | ((td1 & 0xFFFFu) << 16));
  g1[3] = (int)((td1 >> 16) | (tile0 << 16));       // tile_dim0 @ 127:112
  g1[4] = (int)(tile1 & 0xFFFFu);                   // tile_dim1; tile_dim2=0
  g1[5] = (int)(stride0 & 0xFFFFFFFFull);           // tensor_dim0_stride[31:0]
  g1[6] = (int)((stride0 >> 32) & 0xFFFFull);       // stride0[47:32]; dim1_stride[15:0]=0
  g1[7] = 0;
  i32x4 gz = {0, 0, 0, 0};
#if defined(__clang_major__) && (__clang_major__ >= 23)
  i32x8 gz8 = {0, 0, 0, 0, 0, 0, 0, 0};
  __builtin_amdgcn_tensor_load_to_lds(g0, g1, gz, gz, gz8, 0);
#else
  __builtin_amdgcn_tensor_load_to_lds(g0, g1, gz, gz, 0);
#endif
}

// ---------------- kernel 1: zero counters ----------------
__global__ void k_zero(int* counts, int* active) {
  int t = threadIdx.x;
  if (t < NEXP)      counts[t] = 0;
  if (t < 2 * NEXP)  active[t] = 0;
}

// ---------------- kernel 2: router (one wave32 per token) ----------------
__global__ void k_router(const float* __restrict__ x, const float* __restrict__ wg,
                         const float* __restrict__ bg,
                         int* counts, int* active, int* topki, float* topks,
                         int* lists) {
  const int wid  = threadIdx.x >> 5;
  const int lane = threadIdx.x & 31;
  const int t    = blockIdx.x * 8 + wid;

  float acc[NEXP];
#pragma unroll
  for (int e = 0; e < NEXP; e++) acc[e] = 0.f;

  const float* xr = x + (size_t)t * DIM;
  for (int d = lane; d < DIM; d += 32) {
    float xv = xr[d];
    const float* wr = wg + (size_t)d * NEXP;
#pragma unroll
    for (int e = 0; e < NEXP; e++) acc[e] += xv * wr[e];
  }
#pragma unroll
  for (int off = 16; off > 0; off >>= 1) {
#pragma unroll
    for (int e = 0; e < NEXP; e++) acc[e] += __shfl_xor(acc[e], off, 32);
  }

  if (lane == 0) {
    float sc[NEXP], mx = -1e30f;
#pragma unroll
    for (int e = 0; e < NEXP; e++) { sc[e] = acc[e] + bg[e]; mx = sc[e] > mx ? sc[e] : mx; }
    float p[NEXP], sum = 0.f;
#pragma unroll
    for (int e = 0; e < NEXP; e++) { p[e] = __expf(sc[e] - mx); sum += p[e]; }
    float inv = 1.f / sum;
#pragma unroll
    for (int e = 0; e < NEXP; e++) p[e] *= inv;

    int e0 = 0;
#pragma unroll
    for (int e = 1; e < NEXP; e++) if (p[e] > p[e0]) e0 = e;   // strict >: lowest idx on tie
    int e1 = (e0 == 0) ? 1 : 0;
#pragma unroll
    for (int e = 0; e < NEXP; e++) if (e != e0 && p[e] > p[e1]) e1 = e;

    topki[t * 2 + 0] = e0;  topki[t * 2 + 1] = e1;
    topks[t * 2 + 0] = p[e0]; topks[t * 2 + 1] = p[e1];
    active[0 * NEXP + e0] = 1;
    active[1 * NEXP + e1] = 1;
    int pos0 = atomicAdd(&counts[e0], 1);
    lists[e0 * CAP + pos0] = (t << 1) | 0;
    int pos1 = atomicAdd(&counts[e1], 1);
    lists[e1 * CAP + pos1] = (t << 1) | 1;
  }
}

// ---------------- kernel 3: c_e = relu(b1_e) @ W2_e + b2_e ----------------
__global__ void k_cexp(const float* __restrict__ b1, const float* __restrict__ W2,
                       const float* __restrict__ b2, float* __restrict__ cexp) {
  int e = blockIdx.x >> 2;
  int d = ((blockIdx.x & 3) << 8) + threadIdx.x;
  const float* w2e = W2 + (size_t)e * DFF * DIM;
  const float* b1e = b1 + (size_t)e * DFF;
  float s = 0.f;
  for (int f = 0; f < DFF; f++) {
    float rb = b1e[f];
    rb = rb > 0.f ? rb : 0.f;
    s += rb * w2e[(size_t)f * DIM + d];
  }
  cexp[e * DIM + d] = s + b2[e * DIM + d];
}

// ---------------- kernel 4: C_i = sum over active experts of c_e ----------------
__global__ void k_cw(const float* __restrict__ cexp, const int* __restrict__ active,
                     float* __restrict__ Cw) {
  int i = blockIdx.x >> 2;
  int d = ((blockIdx.x & 3) << 8) + threadIdx.x;
  float s = 0.f;
#pragma unroll
  for (int e = 0; e < NEXP; e++)
    if (active[i * NEXP + e]) s += cexp[e * DIM + d];
  Cw[i * DIM + d] = s;
}

// ------- kernel 5: out[t][d] = sum_i s_ti * (C_i[d] - c_{e_ti}[d]) -------
__global__ void k_outinit(const int* __restrict__ topki, const float* __restrict__ topks,
                          const float* __restrict__ cexp, const float* __restrict__ Cw,
                          float* __restrict__ out) {
  int idx = blockIdx.x * 256 + threadIdx.x;
  int t = idx >> 10, d = idx & (DIM - 1);
  int   e0 = topki[t * 2], e1 = topki[t * 2 + 1];
  float s0 = topks[t * 2], s1 = topks[t * 2 + 1];
  out[idx] = s0 * (Cw[d] - cexp[e0 * DIM + d]) + s1 * (Cw[DIM + d] - cexp[e1 * DIM + d]);
}

// ---------------- preconversion kernels (big-ws path) ----------------
__global__ void k_cvt_x(const float* __restrict__ x, unsigned short* __restrict__ xb) {
  int idx = blockIdx.x * 256 + threadIdx.x;
  xb[idx] = f2bf(x[idx]);
}

// in: [NEXP][R][C] fp32  ->  out: [NEXP][C][R] bf16 (LDS-tiled 64x64 transpose)
__global__ void k_transpose_cvt(const float* __restrict__ in,
                                unsigned short* __restrict__ out, int R, int C) {
  __shared__ unsigned short t[64 * 65];
  int tilesR = R >> 6, tilesC = C >> 6;
  int tilesPerE = tilesR * tilesC;
  int e  = blockIdx.x / tilesPerE;
  int tr = (blockIdx.x % tilesPerE) / tilesC;
  int tc = (blockIdx.x % tilesPerE) % tilesC;
  int R0 = tr << 6, C0 = tc << 6;
  const float* ine = in + (size_t)e * R * C;
  unsigned short* oute = out + (size_t)e * C * R;
  for (int idx = threadIdx.x; idx < 64 * 64; idx += 256) {
    int r = idx >> 6, c = idx & 63;
    t[c * 65 + r] = f2bf(ine[(size_t)(R0 + r) * C + C0 + c]);
  }
  __syncthreads();
  for (int idx = threadIdx.x; idx < 64 * 64; idx += 256) {
    int c = idx >> 6, r = idx & 63;
    oute[(size_t)(C0 + c) * R + R0 + r] = t[c * 65 + r];
  }
}

// ---------------- fused grouped FFN GEMM (shared tile geometry) ----------------
#define XS_PITCH  1032
#define W1S_PITCH 264
#define HS_PITCH  136
#define W2S_PITCH 40
#define SMEM_USH  (16 * XS_PITCH + 128 * W1S_PITCH + 16 * HS_PITCH + 1024 * W2S_PITCH)

// ---- variant A (fallback, small ws): fp32 weights, convert while staging ----
__global__ __launch_bounds__(256, 1)
void k_ffn(const float* __restrict__ x, const float* __restrict__ W1,
           const float* __restrict__ b1, const float* __restrict__ W2,
           const float* __restrict__ b2, const int* __restrict__ counts,
           const int* __restrict__ lists, const float* __restrict__ topks,
           float* __restrict__ out) {
  int bid = blockIdx.x;
  int e = -1, te = 0, accT = 0;
#pragma unroll
  for (int i = 0; i < NEXP; i++) {
    int nt = (counts[i] + 15) >> 4;
    if (e < 0 && bid < accT + nt) { e = i; te = bid - accT; }
    accT += nt;
  }
  if (e < 0) return;

  const int tid  = threadIdx.x;
  const int w    = tid >> 5;
  const int lane = tid & 31;
  const int row  = lane & 15;
  const int h    = lane >> 4;
  const int nn   = lane & 15;

  extern __shared__ unsigned short sm[];
  unsigned short* Xs  = sm;
  unsigned short* W1s = Xs + 16 * XS_PITCH;
  unsigned short* Hs  = W1s + 128 * W1S_PITCH;
  unsigned short* W2s = Hs + 16 * HS_PITCH;

  __shared__ int   tokS[16];
  __shared__ float sS[16];
  if (tid < 16) {
    int g = te * 16 + tid;
    if (g < counts[e]) {
      int ent  = lists[e * CAP + g];
      tokS[tid] = ent >> 1;
      sS[tid]   = topks[(ent >> 1) * 2 + (ent & 1)];
    } else { tokS[tid] = -1; sS[tid] = 0.f; }
  }
  __syncthreads();

  for (int idx = tid; idx < 16 * DIM; idx += 256) {
    int r = idx >> 10, c = idx & (DIM - 1);
    int tok = tokS[r];
    float v = (tok >= 0) ? x[(size_t)tok * DIM + c] : 0.f;
    Xs[r * XS_PITCH + c] = f2bf(v);
  }
  __syncthreads();

  const float* W1e = W1 + (size_t)e * DIM * DFF;
  const float* W2e = W2 + (size_t)e * DFF * DIM;
  const float* b1e = b1 + (size_t)e * DFF;
  const float* b2e = b2 + (size_t)e * DIM;

  f32x8 acc[8];
#pragma unroll
  for (int ct = 0; ct < 8; ct++)
    acc[ct] = (f32x8){0.f, 0.f, 0.f, 0.f, 0.f, 0.f, 0.f, 0.f};

  for (int ffc = 0; ffc < DFF; ffc += 128) {
    f32x8 hacc = (f32x8){0.f, 0.f, 0.f, 0.f, 0.f, 0.f, 0.f, 0.f};
    for (int kb0 = 0; kb0 < DIM; kb0 += 256) {
      for (int idx = tid; idx < 256 * 128; idx += 256) {
        int kl = idx >> 7, ffl = idx & 127;
        W1s[ffl * W1S_PITCH + kl] = f2bf(W1e[(size_t)(kb0 + kl) * DFF + ffc + ffl]);
      }
      __syncthreads();
#pragma unroll
      for (int kk = 0; kk < 256; kk += 32) {
        FragAB A, B;
        A.q[0] = *(const uint4*)&Xs[row * XS_PITCH + kb0 + kk + h * 16];
        A.q[1] = *(const uint4*)&Xs[row * XS_PITCH + kb0 + kk + h * 16 + 8];
        const unsigned short* bp = &W1s[(w * 16 + nn) * W1S_PITCH + kk + h * 16];
        B.q[0] = *(const uint4*)(bp);
        B.q[1] = *(const uint4*)(bp + 8);
        hacc = __builtin_amdgcn_wmma_f32_16x16x32_bf16(
            false, A.v, false, B.v, (short)0, hacc, false, false);
      }
      __syncthreads();
    }
    {
      float bias = b1e[ffc + w * 16 + nn];
#pragma unroll
      for (int r = 0; r < 8; r++) {
        float v = hacc[r] + bias;
        v = v > 0.f ? v : 0.f;
        Hs[(r + 8 * h) * HS_PITCH + w * 16 + nn] = f2bf(v);
      }
    }
    __syncthreads();

    for (int ks = 0; ks < 128; ks += 32) {
      for (int idx = tid; idx < 32 * 1024; idx += 256) {
        int kl = idx >> 10, col = idx & (DIM - 1);
        W2s[col * W2S_PITCH + kl] = f2bf(W2e[(size_t)(ffc + ks + kl) * DIM + col]);
      }
      __syncthreads();
      FragAB A;
      A.q[0] = *(const uint4*)&Hs[row * HS_PITCH + ks + h * 16];
      A.q[1] = *(const uint4*)&Hs[row * HS_PITCH + ks + h * 16 + 8];
#pragma unroll
      for (int ct = 0; ct < 8; ct++) {
        FragAB B;
        const unsigned short* bp = &W2s[(w * 128 + ct * 16 + nn) * W2S_PITCH + h * 16];
        B.q[0] = *(const uint4*)(bp);
        B.q[1] = *(const uint4*)(bp + 8);
        acc[ct] = __builtin_amdgcn_wmma_f32_16x16x32_bf16(
            false, A.v, false, B.v, (short)0, acc[ct], false, false);
      }
      __syncthreads();
    }
  }

#pragma unroll
  for (int ct = 0; ct < 8; ct++) {
    int col = w * 128 + ct * 16 + nn;
    float bb = b2e[col];
#pragma unroll
    for (int r = 0; r < 8; r++) {
      int m = r + 8 * h;
      int tok = tokS[m];
      if (tok >= 0)
        atomicAdd(&out[(size_t)tok * DIM + col], sS[m] * (acc[ct][r] + bb));
    }
  }
}

// ---- variant B (big ws): bf16 pre-transposed weights, TDM staging ----
__global__ __launch_bounds__(256, 1)
void k_ffn_bf16(const unsigned short* __restrict__ xb,
                const unsigned short* __restrict__ W1T,   // [e][ff][k] bf16
                const unsigned short* __restrict__ W2T,   // [e][col][ff] bf16
                const float* __restrict__ b1, const float* __restrict__ b2,
                const int* __restrict__ counts, const int* __restrict__ lists,
                const float* __restrict__ topks, float* __restrict__ out) {
  int bid = blockIdx.x;
  int e = -1, te = 0, accT = 0;
#pragma unroll
  for (int i = 0; i < NEXP; i++) {
    int nt = (counts[i] + 15) >> 4;
    if (e < 0 && bid < accT + nt) { e = i; te = bid - accT; }
    accT += nt;
  }
  if (e < 0) return;

  const int tid  = threadIdx.x;
  const int w    = tid >> 5;
  const int lane = tid & 31;
  const int row  = lane & 15;
  const int h    = lane >> 4;
  const int nn   = lane & 15;

  extern __shared__ unsigned short sm[];
  unsigned short* Xs  = sm;
  unsigned short* W1s = Xs + 16 * XS_PITCH;
  unsigned short* Hs  = W1s + 128 * W1S_PITCH;
  unsigned short* W2s = Hs + 16 * HS_PITCH;
  const unsigned w1s_off = (unsigned)(size_t)(void*)W1s;  // LDS byte offsets for D#
  const unsigned w2s_off = (unsigned)(size_t)(void*)W2s;

  __shared__ int   tokS[16];
  __shared__ float sS[16];
  if (tid < 16) {
    int g = te * 16 + tid;
    if (g < counts[e]) {
      int ent  = lists[e * CAP + g];
      tokS[tid] = ent >> 1;
      sS[tid]   = topks[(ent >> 1) * 2 + (ent & 1)];
    } else { tokS[tid] = -1; sS[tid] = 0.f; }
  }
  __syncthreads();

  // X tile: plain 16B copies from pre-converted bf16 x
  for (int idx = tid; idx < 16 * (DIM / 8); idx += 256) {
    int r = idx >> 7, c = idx & 127;
    uint4 v = {0u, 0u, 0u, 0u};
    int tok = tokS[r];
    if (tok >= 0) v = *((const uint4*)(xb + (size_t)tok * DIM) + c);
    *((uint4*)(Xs + r * XS_PITCH) + c) = v;
  }
  __syncthreads();

  const unsigned short* W1Te = W1T + (size_t)e * DFF * DIM;
  const unsigned short* W2Te = W2T + (size_t)e * DIM * DFF;
  const float* b1e = b1 + (size_t)e * DFF;
  const float* b2e = b2 + (size_t)e * DIM;

  f32x8 acc[8];
#pragma unroll
  for (int ct = 0; ct < 8; ct++)
    acc[ct] = (f32x8){0.f, 0.f, 0.f, 0.f, 0.f, 0.f, 0.f, 0.f};

  for (int ffc = 0; ffc < DFF; ffc += 128) {
    f32x8 hacc = (f32x8){0.f, 0.f, 0.f, 0.f, 0.f, 0.f, 0.f, 0.f};
    for (int kb0 = 0; kb0 < DIM; kb0 += 256) {
      if (w == 0) {  // one TDM op stages the whole 128x256 (64 KB) W1 block
        tdm_load_2d(w1s_off, W1Te + (size_t)ffc * DIM + kb0,
                    /*td0=*/DIM, /*td1=*/DFF, /*tile0=*/256, /*tile1=*/128,
                    /*stride0=*/DIM, /*padInterval 128 dw=*/6, /*padAmount 4 dw=*/3);
        __builtin_amdgcn_s_wait_tensorcnt(0);
      }
      __syncthreads();
#pragma unroll
      for (int kk = 0; kk < 256; kk += 32) {
        FragAB A, B;
        A.q[0] = *(const uint4*)&Xs[row * XS_PITCH + kb0 + kk + h * 16];
        A.q[1] = *(const uint4*)&Xs[row * XS_PITCH + kb0 + kk + h * 16 + 8];
        const unsigned short* bp = &W1s[(w * 16 + nn) * W1S_PITCH + kk + h * 16];
        B.q[0] = *(const uint4*)(bp);
        B.q[1] = *(const uint4*)(bp + 8);
        hacc = __builtin_amdgcn_wmma_f32_16x16x32_bf16(
            false, A.v, false, B.v, (short)0, hacc, false, false);
      }
      __syncthreads();
    }
    {
      float bias = b1e[ffc + w * 16 + nn];
#pragma unroll
      for (int r = 0; r < 8; r++) {
        float v = hacc[r] + bias;
        v = v > 0.f ? v : 0.f;
        Hs[(r + 8 * h) * HS_PITCH + w * 16 + nn] = f2bf(v);
      }
    }
    __syncthreads();

    for (int ks = 0; ks < 128; ks += 32) {
      if (w == 0) {  // one TDM op stages the whole 1024x32 (64 KB) W2 block
        tdm_load_2d(w2s_off, W2Te + ffc + ks,
                    /*td0=*/DFF, /*td1=*/DIM, /*tile0=*/32, /*tile1=*/1024,
                    /*stride0=*/DFF, /*padInterval 16 dw=*/3, /*padAmount 4 dw=*/3);
        __builtin_amdgcn_s_wait_tensorcnt(0);
      }
      __syncthreads();
      FragAB A;
      A.q[0] = *(const uint4*)&Hs[row * HS_PITCH + ks + h * 16];
      A.q[1] = *(const uint4*)&Hs[row * HS_PITCH + ks + h * 16 + 8];
#pragma unroll
      for (int ct = 0; ct < 8; ct++) {
        FragAB B;
        const unsigned short* bp = &W2s[(w * 128 + ct * 16 + nn) * W2S_PITCH + h * 16];
        B.q[0] = *(const uint4*)(bp);
        B.q[1] = *(const uint4*)(bp + 8);
        acc[ct] = __builtin_amdgcn_wmma_f32_16x16x32_bf16(
            false, A.v, false, B.v, (short)0, acc[ct], false, false);
      }
      __syncthreads();
    }
  }

#pragma unroll
  for (int ct = 0; ct < 8; ct++) {
    int col = w * 128 + ct * 16 + nn;
    float bb = b2e[col];
#pragma unroll
    for (int r = 0; r < 8; r++) {
      int m = r + 8 * h;
      int tok = tokS[m];
      if (tok >= 0)
        atomicAdd(&out[(size_t)tok * DIM + col], sS[m] * (acc[ct][r] + bb));
    }
  }
}

// ---------------- host launcher ----------------
extern "C" void kernel_launch(void* const* d_in, const int* in_sizes, int n_in,
                              void* d_out, int out_size, void* d_ws, size_t ws_size,
                              hipStream_t stream) {
  const float* x  = (const float*)d_in[0];
  const float* wg = (const float*)d_in[1];
  const float* bg = (const float*)d_in[2];
  const float* W1 = (const float*)d_in[3];
  const float* b1 = (const float*)d_in[4];
  const float* W2 = (const float*)d_in[5];
  const float* b2 = (const float*)d_in[6];
  float* out = (float*)d_out;

  char* ws = (char*)d_ws;
  int*   counts = (int*)(ws + 0);
  int*   active = (int*)(ws + 32);
  int*   topki  = (int*)(ws + 128);
  float* topks  = (float*)(ws + 128 + 131072);
  float* cexp   = (float*)(ws + 128 + 262144);
  float* Cw     = (float*)(ws + 128 + 262144 + 32768);
  int*   lists  = (int*)(ws + 128 + 262144 + 32768 + 8192);

  const size_t MB = 1024ull * 1024ull;
  const size_t xbf_off = 2 * MB;
  const size_t w1t_off = xbf_off + 32 * MB;
  const size_t w2t_off = w1t_off + 64 * MB;
  const size_t req     = w2t_off + 64 * MB;      // 162 MB for the bf16+TDM path
  const bool big = (ws_size >= req);

  k_zero   <<<1, 32, 0, stream>>>(counts, active);
  k_router <<<N_TOK / 8, 256, 0, stream>>>(x, wg, bg, counts, active, topki, topks, lists);
  k_cexp   <<<NEXP * 4, 256, 0, stream>>>(b1, W2, b2, cexp);
  k_cw     <<<2 * 4, 256, 0, stream>>>(cexp, active, Cw);
  k_outinit<<<(N_TOK * DIM) / 256, 256, 0, stream>>>(topki, topks, cexp, Cw, out);

  const size_t smem = (size_t)SMEM_USH * sizeof(unsigned short);  // ~183 KB dynamic LDS
  if (big) {
    unsigned short* xbf = (unsigned short*)(ws + xbf_off);
    unsigned short* W1T = (unsigned short*)(ws + w1t_off);
    unsigned short* W2T = (unsigned short*)(ws + w2t_off);
    k_cvt_x<<<(N_TOK * DIM) / 256, 256, 0, stream>>>(x, xbf);
    k_transpose_cvt<<<NEXP * (DIM / 64) * (DFF / 64), 256, 0, stream>>>(W1, W1T, DIM, DFF);
    k_transpose_cvt<<<NEXP * (DFF / 64) * (DIM / 64), 256, 0, stream>>>(W2, W2T, DFF, DIM);
    k_ffn_bf16<<<2056, 256, smem, stream>>>(xbf, W1T, W2T, b1, b2, counts, lists, topks, out);
  } else {
    k_ffn<<<2056, 256, smem, stream>>>(x, W1, b1, W2, b2, counts, lists, topks, out);
  }
}
